// CrystalAE_13116830122572
// MI455X (gfx1250) — matrix operand order repaired and checked by reading, not compiled, not run on hardware
//
#include <hip/hip_runtime.h>
#include <hip/hip_bf16.h>
#include <math.h>

// ---------------------------------------------------------------------------
// Problem constants (match reference)
// ---------------------------------------------------------------------------
#define N_ATOMS 24576        // B*NA = 512*48
#define ORIG    92
#define NBR     41
#define DIM     64
#define MNB     12
#define NCONV   3
#define BATCH   512
#define NA      48
#define ROWS    (N_ATOMS*MNB)   // 294912
#define KCONV   192             // 2D+NBR = 169 padded to 6*32
#define KEMB    96              // 92 padded to 3*32
#define TMPC    704             // (6+5)*64 bilinear temp columns
#define NDEC    800             // 704 + 96 (atom head, 92 padded)
#define EPSBN   1e-5f

#define EDGE_P_SZ (BATCH*NA*NA*6)
#define ATOM_SZ   (BATCH*NA*ORIG)
#define OFF_ATOM  (EDGE_P_SZ)
#define OFF_EDGEF (EDGE_P_SZ + ATOM_SZ)

typedef __attribute__((ext_vector_type(16))) _Float16 v16h;
typedef __attribute__((ext_vector_type(8)))  float    v8f;

// ---------------------------------------------------------------------------
// WMMA fragment helpers (wave32, 16x16x32 f16, A/B in contiguous K order, C per ISA)
// A 16x32: lanes 0-15 -> M=lane, elems K0..15; lanes 16-31 -> K16..31
// B 32x16: lanes 0-15 -> N=lane, elems K0..15; lanes 16-31 -> K16..31
// C 16x16: elem j: lanes 0-15 -> M=j, N=lane; lanes 16-31 -> M=j+8, N=lane-16
// All panels are stored N-major (Bt[n*ldb + k]) so every lane's 16 K-values are
// contiguous -> two 16B loads per fragment, no packing movs.
// ---------------------------------------------------------------------------
__device__ __forceinline__ v16h ldA(const _Float16* tile, int lda, int kt, int lane) {
    int m  = lane & 15;
    int kb = kt * 32 + ((lane & 16) ? 16 : 0);
    const _Float16* r = tile + m * lda + kb;
    v16h a;
#pragma unroll
    for (int i = 0; i < 16; ++i) a[i] = r[i];
    return a;
}

// B stored N-major: Bt[n*ldb + k] -> contiguous 16 halves per lane
__device__ __forceinline__ v16h ldBT(const _Float16* Bt, int ldb, int kt, int n0, int lane) {
    int n  = n0 + (lane & 15);
    int kb = kt * 32 + ((lane & 16) ? 16 : 0);
    const _Float16* r = Bt + n * ldb + kb;
    v16h b;
#pragma unroll
    for (int i = 0; i < 16; ++i) b[i] = r[i];
    return b;
}

__device__ __forceinline__ v8f wmma32(v16h a, v16h b, v8f c) {
    return __builtin_amdgcn_wmma_f32_16x16x32_f16(false, a, false, b, (short)0, c, false, false);
}

__device__ __forceinline__ float softplusf_(float x) { return x > 20.f ? x : log1pf(expf(x)); }
__device__ __forceinline__ float sigmoidf_(float x)  { return 1.f / (1.f + expf(-x)); }

// ---------------------------------------------------------------------------
// Weight conversion kernels (f32 -> f16, N-major, zero-padded K)
// ---------------------------------------------------------------------------
// W_embN[d][k], k padded 92->96
__global__ void k_cvt_emb(const float* __restrict__ W_emb, _Float16* __restrict__ W_embN) {
    for (int idx = blockIdx.x * blockDim.x + threadIdx.x; idx < DIM * KEMB;
         idx += gridDim.x * blockDim.x) {
        int d = idx / KEMB, k = idx % KEMB;
        W_embN[idx] = (k < ORIG) ? (_Float16)W_emb[d * ORIG + k] : (_Float16)0.f;
    }
}

// Wt[l][c][k], k padded 169->192 (same row order as W_full)
__global__ void k_cvt_conv(const float* __restrict__ W_full, _Float16* __restrict__ Wt) {
    for (int idx = blockIdx.x * blockDim.x + threadIdx.x; idx < NCONV * 128 * KCONV;
         idx += gridDim.x * blockDim.x) {
        int lc = idx / KCONV;           // l*128 + c
        int k  = idx % KCONV;
        Wt[idx] = (k < 2 * DIM + NBR) ? (_Float16)W_full[lc * (2 * DIM + NBR) + k]
                                      : (_Float16)0.f;
    }
}

// Bcat[j][d], j in [0,800): 6*64 adj cols, 5*64 edge cols, 96 atom cols (92 used)
__global__ void k_cvt_dec(const float* __restrict__ W_adj, const float* __restrict__ W_edge,
                          const float* __restrict__ W_atom, _Float16* __restrict__ Bcat) {
    for (int idx = blockIdx.x * blockDim.x + threadIdx.x; idx < NDEC * DIM;
         idx += gridDim.x * blockDim.x) {
        int j = idx / DIM, d = idx % DIM;
        float v;
        if (j < 384) {                       // adj: Bcat[k*64+e][d] = W_adj[k][d][e]
            int k = j >> 6, e = j & 63;
            v = W_adj[(k * DIM + d) * DIM + e];
        } else if (j < TMPC) {               // edge bilinear mats
            int jj = j - 384;
            int k = jj >> 6, e = jj & 63;
            v = W_edge[(k * DIM + d) * DIM + e];
        } else {                             // atom head: Bcat[704+o][d] = W_atom[o][d]
            int o = j - TMPC;
            v = (o < ORIG) ? W_atom[o * DIM + d] : 0.f;
        }
        Bcat[idx] = (_Float16)v;
    }
}

// ---------------------------------------------------------------------------
// Embedding GEMM: fea = atom_fea @ W_emb^T   [24576x92]x[92x64]
// grid = N/16 blocks, 128 threads (4 waves = 4 N-tiles)
// ---------------------------------------------------------------------------
__global__ void k_embed(const float* __restrict__ atom_fea, const _Float16* __restrict__ W_embN,
                        float* __restrict__ fea, _Float16* __restrict__ fea_h) {
    __shared__ _Float16 As[16 * KEMB];
    int tid = threadIdx.x, lane = tid & 31, wave = tid >> 5;
    int rowBase = blockIdx.x * 16;
    for (int idx = tid; idx < 16 * KEMB; idx += 128) {
        int r = idx / KEMB, c = idx % KEMB;
        As[idx] = (c < ORIG) ? (_Float16)atom_fea[(rowBase + r) * ORIG + c] : (_Float16)0.f;
    }
    __syncthreads();
    v8f acc = {};
#pragma unroll
    for (int kt = 0; kt < 3; ++kt)
        acc = wmma32(ldA(As, KEMB, kt, lane), ldBT(W_embN, KEMB, kt, wave * 16, lane), acc);
    int n  = wave * 16 + (lane & 15);
    int mb = (lane & 16) ? 8 : 0;
#pragma unroll
    for (int j = 0; j < 8; ++j) {
        int row = rowBase + mb + j;
        fea[row * DIM + n]   = acc[j];
        fea_h[row * DIM + n] = (_Float16)acc[j];
    }
}

// ---------------------------------------------------------------------------
// BN statistics buffers: stats[0..127]=sum1, [128..255]=sq1, [256..319]=sum2,
// [320..383]=sq2
// ---------------------------------------------------------------------------
__global__ void k_bnzero(float* __restrict__ stats) {
    int t = blockIdx.x * blockDim.x + threadIdx.x;
    if (t < 384) stats[t] = 0.f;
}

// ---------------------------------------------------------------------------
// Conv GEMM: gated = concat(self, nbr, edge) @ W^T + b  [294912 x 169]x[169 x 128]
// grid = ROWS/16 blocks, 256 threads (8 waves = 8 N-tiles of 16 -> 128 cols)
// A tile built in LDS from fea_h gather + nbr_fea (f32->f16), K padded 169->192.
// ---------------------------------------------------------------------------
__global__ void k_conv(const _Float16* __restrict__ fea_h, const int* __restrict__ nbr_idx,
                       const float* __restrict__ nbr_fea, const _Float16* __restrict__ Wt_l,
                       const float* __restrict__ b_l, float* __restrict__ gated) {
    __shared__ _Float16 As[16 * KCONV];
    int tid = threadIdx.x, lane = tid & 31, wave = tid >> 5;
    int rowBase = blockIdx.x * 16;
    for (int idx = tid; idx < 16 * KCONV; idx += 256) {
        int r = idx / KCONV, c = idx % KCONV;
        int pair = rowBase + r;
        int i = pair / MNB, m = pair - i * MNB;
        _Float16 v;
        if (c < DIM)            v = fea_h[i * DIM + c];
        else if (c < 2 * DIM)   v = fea_h[nbr_idx[i * MNB + m] * DIM + (c - DIM)];
        else if (c < 2 * DIM + NBR) v = (_Float16)nbr_fea[(i * MNB + m) * NBR + (c - 2 * DIM)];
        else                    v = (_Float16)0.f;
        As[idx] = v;
    }
    __syncthreads();
    v8f acc = {};
#pragma unroll
    for (int kt = 0; kt < 6; ++kt)
        acc = wmma32(ldA(As, KCONV, kt, lane), ldBT(Wt_l, KCONV, kt, wave * 16, lane), acc);
    int col = wave * 16 + (lane & 15);
    int mb  = (lane & 16) ? 8 : 0;
    float bias = b_l[col];
#pragma unroll
    for (int j = 0; j < 8; ++j)
        gated[(rowBase + mb + j) * 128 + col] = acc[j] + bias;
}

// BN1 stats: per-channel sum/sumsq over ROWS rows (thread-local then atomic)
__global__ void k_bn1(const float* __restrict__ gated, float* __restrict__ stats) {
    int c = threadIdx.x;                 // 128 channels
    float s = 0.f, q = 0.f;
    for (int r = blockIdx.x; r < ROWS; r += gridDim.x) {
        float v = gated[r * 128 + c];
        s += v; q += v * v;
    }
    atomicAdd(&stats[c], s);
    atomicAdd(&stats[128 + c], q);
}

// Gate + neighbor sum: summed[i,c] = sum_m sigmoid(bn(filt))*softplus(bn(core));
// also accumulates BN2 statistics (stats is read for BN1, atomically written
// for BN2 -> must be non-const).
__global__ void k_gate(const float* __restrict__ gated, float* __restrict__ stats,
                       const float* __restrict__ g1_l, const float* __restrict__ be1_l,
                       float* __restrict__ summed) {
    int t = blockIdx.x * blockDim.x + threadIdx.x;
    if (t >= N_ATOMS * DIM) return;
    int i = t / DIM, c = t & 63;
    const float inv = 1.f / (float)ROWS;
    float muF = stats[c] * inv;
    float vaF = stats[128 + c] * inv - muF * muF;
    float scF = rsqrtf(vaF + EPSBN) * g1_l[c];
    float shF = be1_l[c] - muF * scF;
    int cc = c + DIM;
    float muC = stats[cc] * inv;
    float vaC = stats[128 + cc] * inv - muC * muC;
    float scC = rsqrtf(vaC + EPSBN) * g1_l[cc];
    float shC = be1_l[cc] - muC * scC;
    float acc = 0.f;
#pragma unroll
    for (int m = 0; m < MNB; ++m) {
        const float* row = gated + (size_t)(i * MNB + m) * 128;
        float xf = row[c]  * scF + shF;
        float xc = row[cc] * scC + shC;
        acc += sigmoidf_(xf) * softplusf_(xc);
    }
    summed[t] = acc;
    atomicAdd(&stats[256 + c], acc);
    atomicAdd(&stats[320 + c], acc * acc);
}

// fea = softplus(fea + bn2(summed))
__global__ void k_update(const float* __restrict__ summed, const float* __restrict__ stats,
                         const float* __restrict__ g2_l, const float* __restrict__ be2_l,
                         float* __restrict__ fea, _Float16* __restrict__ fea_h) {
    int t = blockIdx.x * blockDim.x + threadIdx.x;
    if (t >= N_ATOMS * DIM) return;
    int c = t & 63;
    const float inv = 1.f / (float)N_ATOMS;
    float mu = stats[256 + c] * inv;
    float va = stats[320 + c] * inv - mu * mu;
    float x  = (summed[t] - mu) * rsqrtf(va + EPSBN) * g2_l[c] + be2_l[c];
    float f  = softplusf_(fea[t] + x);
    fea[t]   = f;
    fea_h[t] = (_Float16)f;
}

// ---------------------------------------------------------------------------
// Decode stage-1 GEMM: [24576x64] x [64x800] -> bilinear temps (f16) + atom head
// grid = N/16 blocks, 256 threads (8 waves loop over 50 N-tiles)
// ---------------------------------------------------------------------------
__global__ void k_decode(const _Float16* __restrict__ fea_h, const _Float16* __restrict__ Bcat,
                         const float* __restrict__ b_atom, _Float16* __restrict__ tmp_h,
                         float* __restrict__ out) {
    __shared__ _Float16 As[16 * DIM];
    int tid = threadIdx.x, lane = tid & 31, wave = tid >> 5;
    int rowBase = blockIdx.x * 16;
    for (int idx = tid; idx < 16 * DIM; idx += 256)
        As[idx] = fea_h[(rowBase + idx / DIM) * DIM + (idx % DIM)];
    __syncthreads();
    for (int nt = wave; nt < NDEC / 16; nt += 8) {
        v8f acc = {};
#pragma unroll
        for (int kt = 0; kt < 2; ++kt)
            acc = wmma32(ldA(As, DIM, kt, lane), ldBT(Bcat, DIM, kt, nt * 16, lane), acc);
        int col = nt * 16 + (lane & 15);
        int mb  = (lane & 16) ? 8 : 0;
        if (col < TMPC) {
#pragma unroll
            for (int j = 0; j < 8; ++j)
                tmp_h[(size_t)(rowBase + mb + j) * TMPC + col] = (_Float16)acc[j];
        } else {
            int o = col - TMPC;
            if (o < ORIG) {
                float bias = b_atom[o];
#pragma unroll
                for (int j = 0; j < 8; ++j)
                    out[OFF_ATOM + (size_t)(rowBase + mb + j) * ORIG + o] = acc[j] + bias;
            }
        }
    }
}

// ---------------------------------------------------------------------------
// Per-crystal bilinear decode + fc1/fc2 + log_softmax.
// One block per crystal (512), 256 threads. 48x48x11 channel slab in LDS (f16).
// out[b,m,n,k] = sum_e tmp[b,m,k,e] * bt[b,n,e]
// ---------------------------------------------------------------------------
__global__ void k_edge(const _Float16* __restrict__ fea_h, const _Float16* __restrict__ tmp_h,
                       const int* __restrict__ cidx, const float* __restrict__ b_adj,
                       const float* __restrict__ W_fc1, const float* __restrict__ b_fc1,
                       const float* __restrict__ b_edge, const float* __restrict__ W_fc2,
                       const float* __restrict__ b_fc2, float* __restrict__ out) {
    __shared__ _Float16 Bt[NA * DIM];        // bt, N-major  (6 KB)
    __shared__ _Float16 Aq[NA * DIM];        // tmp slab for current channel (6 KB)
    __shared__ _Float16 res[NA * NA * 11];   // result slab (50688 B)
    int tid = threadIdx.x, lane = tid & 31, wave = tid >> 5;
    int b = blockIdx.x;
    for (int idx = tid; idx < NA * DIM; idx += 256) {
        int n = idx / DIM, e = idx % DIM;
        Bt[idx] = fea_h[(size_t)cidx[b * NA + n] * DIM + e];
    }
    for (int k = 0; k < 11; ++k) {
        __syncthreads();
        for (int idx = tid; idx < NA * DIM; idx += 256) {
            int m = idx / DIM, e = idx % DIM;
            Aq[idx] = tmp_h[(size_t)cidx[b * NA + m] * TMPC + k * DIM + e];
        }
        __syncthreads();
        for (int t = wave; t < 9; t += 8) {
            int mt = t / 3, nt = t % 3;
            v8f acc = {};
#pragma unroll
            for (int kt = 0; kt < 2; ++kt)
                acc = wmma32(ldA(Aq + mt * 16 * DIM, DIM, kt, lane),
                             ldBT(Bt, DIM, kt, nt * 16, lane), acc);
            int n  = nt * 16 + (lane & 15);
            int mb = mt * 16 + ((lane & 16) ? 8 : 0);
#pragma unroll
            for (int j = 0; j < 8; ++j)
                res[((mb + j) * NA + n) * 11 + k] = (_Float16)acc[j];
        }
    }
    __syncthreads();
    for (int p = tid; p < NA * NA; p += 256) {
        // edge_p: 6 channels, fc1, log_softmax
        float v6[6], y[6];
#pragma unroll
        for (int k = 0; k < 6; ++k) v6[k] = (float)res[p * 11 + k] + b_adj[k];
        float mx = -1e30f;
#pragma unroll
        for (int j = 0; j < 6; ++j) {
            float a = b_fc1[j];
#pragma unroll
            for (int k = 0; k < 6; ++k) a += v6[k] * W_fc1[j * 6 + k];
            y[j] = a;
            mx = fmaxf(mx, a);
        }
        float se = 0.f;
#pragma unroll
        for (int j = 0; j < 6; ++j) se += expf(y[j] - mx);
        float lse = mx + logf(se);
        float* op = out + (size_t)b * NA * NA * 6 + (size_t)p * 6;
#pragma unroll
        for (int j = 0; j < 6; ++j) op[j] = y[j] - lse;
        // edge_f: 5 channels, fc2
        float v5[5];
#pragma unroll
        for (int k = 0; k < 5; ++k) v5[k] = (float)res[p * 11 + 6 + k] + b_edge[k];
        float* of = out + OFF_EDGEF + (size_t)b * NA * NA * 5 + (size_t)p * 5;
#pragma unroll
        for (int j = 0; j < 5; ++j) {
            float a = b_fc2[j];
#pragma unroll
            for (int k = 0; k < 5; ++k) a += v5[k] * W_fc2[j * 5 + k];
            of[j] = a;
        }
    }
}

// ---------------------------------------------------------------------------
// Host launcher
// ---------------------------------------------------------------------------
extern "C" void kernel_launch(void* const* d_in, const int* in_sizes, int n_in,
                              void* d_out, int out_size, void* d_ws, size_t ws_size,
                              hipStream_t stream) {
    const float* atom_fea = (const float*)d_in[0];
    const float* nbr_fea  = (const float*)d_in[1];
    const int*   nbr_idx  = (const int*)d_in[2];
    const int*   cidx     = (const int*)d_in[3];
    const float* W_emb    = (const float*)d_in[4];
    const float* W_full   = (const float*)d_in[5];
    const float* b_full   = (const float*)d_in[6];
    const float* g1       = (const float*)d_in[7];
    const float* be1      = (const float*)d_in[8];
    const float* g2       = (const float*)d_in[9];
    const float* be2      = (const float*)d_in[10];
    const float* W_adj    = (const float*)d_in[11];
    const float* b_adj    = (const float*)d_in[12];
    const float* W_fc1    = (const float*)d_in[13];
    const float* b_fc1    = (const float*)d_in[14];
    const float* W_edge   = (const float*)d_in[15];
    const float* b_edge   = (const float*)d_in[16];
    const float* W_fc2    = (const float*)d_in[17];
    const float* b_fc2    = (const float*)d_in[18];
    const float* W_atom   = (const float*)d_in[19];
    const float* b_atom   = (const float*)d_in[20];
    float* out = (float*)d_out;

    char* ws = (char*)d_ws;
    size_t off = 0;
    auto take = [&](size_t bytes) {
        size_t r = off;
        off += (bytes + 255) & ~(size_t)255;
        return r;
    };
    float*    fea    = (float*)   (ws + take((size_t)N_ATOMS * DIM * 4));
    _Float16* fea_h  = (_Float16*)(ws + take((size_t)N_ATOMS * DIM * 2));
    float*    summed = (float*)   (ws + take((size_t)N_ATOMS * DIM * 4));
    float*    stats  = (float*)   (ws + take(384 * 4));
    _Float16* W_embN = (_Float16*)(ws + take((size_t)DIM * KEMB * 2));
    _Float16* Wt     = (_Float16*)(ws + take((size_t)NCONV * 128 * KCONV * 2));
    _Float16* Bcat   = (_Float16*)(ws + take((size_t)NDEC * DIM * 2));
    _Float16* tmp_h  = (_Float16*)(ws + take((size_t)N_ATOMS * TMPC * 2));
    float*    gated  = (float*)   (ws + take((size_t)ROWS * 128 * 4));

    // --- weight conversion (cheap, deterministic) ---
    k_cvt_emb <<<24, 256, 0, stream>>>(W_emb, W_embN);
    k_cvt_conv<<<288, 256, 0, stream>>>(W_full, Wt);
    k_cvt_dec <<<200, 256, 0, stream>>>(W_adj, W_edge, W_atom, Bcat);

    // --- embedding GEMM ---
    k_embed<<<N_ATOMS / 16, 128, 0, stream>>>(atom_fea, W_embN, fea, fea_h);

    // --- conv layers ---
    for (int l = 0; l < NCONV; ++l) {
        k_bnzero<<<2, 256, 0, stream>>>(stats);
        k_conv<<<ROWS / 16, 256, 0, stream>>>(fea_h, nbr_idx, nbr_fea,
                                              Wt + (size_t)l * 128 * KCONV,
                                              b_full + l * 128, gated);
        k_bn1<<<512, 128, 0, stream>>>(gated, stats);
        k_gate<<<(N_ATOMS * DIM) / 256, 256, 0, stream>>>(gated, stats, g1 + l * 128,
                                                          be1 + l * 128, summed);
        k_update<<<(N_ATOMS * DIM) / 256, 256, 0, stream>>>(summed, stats, g2 + l * DIM,
                                                            be2 + l * DIM, fea, fea_h);
    }

    // --- decode ---
    k_decode<<<N_ATOMS / 16, 256, 0, stream>>>(fea_h, Bcat, b_atom, tmp_h, out);
    k_edge<<<BATCH, 256, 0, stream>>>(fea_h, tmp_h, cidx, b_adj, W_fc1, b_fc1,
                                      b_edge, W_fc2, b_fc2, out);
}
